// Attention_17377437680437
// MI455X (gfx1250) — compile-verified, matrix operand order checked by reading, not run on hardware
//
#include <hip/hip_runtime.h>

// ---------------------------------------------------------------------------
// MI455X / gfx1250 GQA attention forward.
// v2.1: pre-cast everything to bf16 once; GEMM tiles moved by the Tensor Data
// Mover (tensor_load_to_lds, TENSORcnt-tracked, double-buffered); all matmuls
// via v_wmma_f32_16x16x32_bf16; fp32 online softmax.
// ---------------------------------------------------------------------------

typedef __attribute__((ext_vector_type(16))) __bf16    v16bf;
typedef __attribute__((ext_vector_type(8)))  __bf16    v8bf;
typedef __attribute__((ext_vector_type(8)))  float     v8f;
typedef __attribute__((ext_vector_type(4)))  unsigned  u32x4;
typedef __attribute__((ext_vector_type(8)))  int       i32x8;
typedef __attribute__((ext_vector_type(4)))  int       i32x4;

#define WMMA_BF16(a, b, c) \
  __builtin_amdgcn_wmma_f32_16x16x32_bf16(false, (a), false, (b), (short)0, (c), false, false)

__device__ __forceinline__ __bf16 f2bf(float f) {
  union { float f; unsigned u; } in; in.f = f;
  unsigned r = in.u + 0x7FFFu + ((in.u >> 16) & 1u);   // round-to-nearest-even
  unsigned short h = (unsigned short)(r >> 16);
  return __builtin_bit_cast(__bf16, h);
}

__device__ __forceinline__ v8f vzero8() {
  v8f z = {0.f, 0.f, 0.f, 0.f, 0.f, 0.f, 0.f, 0.f};
  return z;
}

__device__ __forceinline__ v16bf cat8(v8bf lo, v8bf hi) {
  return __builtin_shufflevector(lo, hi, 0,1,2,3,4,5,6,7,8,9,10,11,12,13,14,15);
}

// 16x32 bf16 fragment load (A-layout; B uses the same loader on [n][k] tiles).
__device__ __forceinline__ v16bf frag_ld(const __bf16* rowbase, int hh) {
  v8bf lo = *(const v8bf*)(rowbase + hh * 8);
  v8bf hi = *(const v8bf*)(rowbase + 16 + hh * 8);
  return cat8(lo, hi);
}

// 32-bit LDS byte address (for TDM descriptor group 0)
__device__ __forceinline__ unsigned lds_addr32(void* p) {
  return (unsigned)(size_t)(__attribute__((address_space(3))) char*)p;
}

// ---------------------------------------------------------------------------
// TDM: load a [rows=128] x [cols=32 bf16] tile from a row-major bf16 tensor
// (row stride = K elements) into LDS, with HW padding of 16B per 64B row
// (pad_interval=16 dwords -> code 3, pad_amount=4 dwords -> code 3), giving
// an 80-byte LDS row pitch (= GP bf16 elements).
// ---------------------------------------------------------------------------
__device__ __forceinline__ void tdm_load_tile(unsigned lds, unsigned long long gaddr,
                                              unsigned K) {
  u32x4 g0 = { 1u,                                               // count=1, user mode
               lds,                                              // lds_addr
               (unsigned)gaddr,                                  // global_addr[31:0]
               (unsigned)((gaddr >> 32) & 0x1FFFFFFu) | (2u << 30) };  // addr[56:32] | type=2
  i32x8 g1 = { (int)((1u << 16) | (1u << 20) | (3u << 22) | (3u << 25)),
               //      data_size=2B  pad_en     pad_int=16dw pad_amt=4dw
               (int)((K & 0xFFFFu) << 16),                       // tensor_dim0 lo16
               (int)(((K >> 16) & 0xFFFFu) | (4096u << 16)),     // dim0 hi | dim1 lo
               (int)(32u << 16),                                 // dim1 hi=0 | tile_dim0=32
               (int)128u,                                        // tile_dim1=128, tile_dim2=0
               (int)K,                                           // dim0_stride lo32 (elements)
               0, 0 };                                           // stride hi / dim1_stride
  i32x4 z4 = { 0, 0, 0, 0 };
  i32x8 z8 = { 0, 0, 0, 0, 0, 0, 0, 0 };
  __builtin_amdgcn_tensor_load_to_lds(g0, g1, z4, z4, z8, 0);
}

// ---------------------------------------------------------------------------
// GEMM: C[M,N](fp32) = A[M,K](bf16, row-major) * BT[N,K](bf16, row-major)^T
// Block: 256 threads (8 waves), 128x128 tile, K-step 32, TDM double-buffered.
// Wave grid 4(M) x 2(N): each wave owns 32x64 -> 2x4 v8f accumulators.
// ---------------------------------------------------------------------------
#define GP 40  // LDS row pitch in bf16 (64B data + 16B TDM pad)

__global__ __launch_bounds__(256) void gemm_bf16_tdm(
    const __bf16* __restrict__ A, const __bf16* __restrict__ BT,
    float* __restrict__ C, int K, int N) {
  __shared__ __attribute__((aligned(16))) __bf16 As[2][128 * GP];  // [m][k]
  __shared__ __attribute__((aligned(16))) __bf16 Bs[2][128 * GP];  // [n][k]

  const int tid  = threadIdx.x;
  const int lane = tid & 31;
  const int wave = tid >> 5;
  const int m0   = blockIdx.y * 128;
  const int n0   = blockIdx.x * 128;
  const int wm   = (wave & 3) * 32;
  const int wn   = (wave >> 2) * 64;
  const int hh   = lane >> 4;
  const int nl   = lane & 15;

  v8f acc[2][4];
#pragma unroll
  for (int i = 0; i < 2; ++i)
#pragma unroll
    for (int j = 0; j < 4; ++j) acc[i][j] = vzero8();

  const int nkt = K >> 5;
  // Prologue: DMA first tile pair into buffer 0 (one wave issues; EXEC-uniform branch)
  if (tid < 32) {
    tdm_load_tile(lds_addr32(&As[0][0]), (size_t)(A  + (size_t)m0 * K), (unsigned)K);
    tdm_load_tile(lds_addr32(&Bs[0][0]), (size_t)(BT + (size_t)n0 * K), (unsigned)K);
  }
  __builtin_amdgcn_s_wait_tensorcnt(0);
  __syncthreads();

  for (int kt = 0; kt < nkt; ++kt) {
    const int cur = kt & 1;
    if (kt + 1 < nkt && tid < 32) {   // prefetch next tile pair while computing
      tdm_load_tile(lds_addr32(&As[cur ^ 1][0]),
                    (size_t)(A + (size_t)m0 * K + (kt + 1) * 32), (unsigned)K);
      tdm_load_tile(lds_addr32(&Bs[cur ^ 1][0]),
                    (size_t)(BT + (size_t)n0 * K + (kt + 1) * 32), (unsigned)K);
    }

    const __bf16* Ab = &As[cur][0];
    const __bf16* Bb = &Bs[cur][0];
    v16bf a0 = frag_ld(&Ab[(wm +  0 + nl) * GP], hh);
    v16bf a1 = frag_ld(&Ab[(wm + 16 + nl) * GP], hh);
    v16bf b0 = frag_ld(&Bb[(wn +  0 + nl) * GP], hh);
    v16bf b1 = frag_ld(&Bb[(wn + 16 + nl) * GP], hh);
    v16bf b2 = frag_ld(&Bb[(wn + 32 + nl) * GP], hh);
    v16bf b3 = frag_ld(&Bb[(wn + 48 + nl) * GP], hh);

    acc[0][0] = WMMA_BF16(a0, b0, acc[0][0]);
    acc[0][1] = WMMA_BF16(a0, b1, acc[0][1]);
    acc[0][2] = WMMA_BF16(a0, b2, acc[0][2]);
    acc[0][3] = WMMA_BF16(a0, b3, acc[0][3]);
    acc[1][0] = WMMA_BF16(a1, b0, acc[1][0]);
    acc[1][1] = WMMA_BF16(a1, b1, acc[1][1]);
    acc[1][2] = WMMA_BF16(a1, b2, acc[1][2]);
    acc[1][3] = WMMA_BF16(a1, b3, acc[1][3]);

    __builtin_amdgcn_s_wait_tensorcnt(0);  // next tiles landed (no-op for waves 1..7)
    __syncthreads();
  }

  // C/D layout: lane nl = column, VGPR r = row (r + 8*hh)
#pragma unroll
  for (int i = 0; i < 2; ++i)
#pragma unroll
    for (int j = 0; j < 4; ++j) {
      size_t mb = (size_t)(m0 + wm + 16 * i + 8 * hh);
      int nn = n0 + wn + 16 * j + nl;
#pragma unroll
      for (int r = 0; r < 8; ++r)
        C[(mb + r) * (size_t)N + nn] = acc[i][j][r];
    }
}

// ---------------------------------------------------------------------------
// One-time fp32 -> bf16 cast (same layout)
// ---------------------------------------------------------------------------
__global__ void cast_bf16(const float* __restrict__ s, __bf16* __restrict__ d,
                          unsigned n4) {
  unsigned i = blockIdx.x * blockDim.x + threadIdx.x;
  if (i >= n4) return;
  float4 v = *(const float4*)(s + (size_t)i * 4);
  __bf16* o = d + (size_t)i * 4;
  o[0] = f2bf(v.x); o[1] = f2bf(v.y); o[2] = f2bf(v.z); o[3] = f2bf(v.w);
}

// ---------------------------------------------------------------------------
// One-time transpose + cast: W[K][N] fp32 -> WT[N][K] bf16 (32x32 LDS tiles)
// ---------------------------------------------------------------------------
__global__ __launch_bounds__(256) void transpose_cast(
    const float* __restrict__ W, __bf16* __restrict__ WT, int Kd, int Nd) {
  __shared__ float tile[32][33];
  int k0 = blockIdx.y * 32, n0 = blockIdx.x * 32;
  int tx = threadIdx.x & 31, ty = threadIdx.x >> 5;  // 32 x 8
#pragma unroll
  for (int i = 0; i < 32; i += 8)
    tile[ty + i][tx] = W[(size_t)(k0 + ty + i) * Nd + n0 + tx];
  __syncthreads();
#pragma unroll
  for (int i = 0; i < 32; i += 8)
    WT[(size_t)(n0 + ty + i) * Kd + k0 + tx] = f2bf(tile[tx][ty + i]);
}

// ---------------------------------------------------------------------------
// RoPE + repack: src fp32 [b][t][nh][128] -> dst bf16 [b][nh][t][128]
// ---------------------------------------------------------------------------
__global__ void rope_pack(const float* __restrict__ src, __bf16* __restrict__ dst,
                          int nheads, int do_rope) {
  size_t idx = (size_t)blockIdx.x * blockDim.x + threadIdx.x;
  size_t total = (size_t)2 * 2048 * nheads * 64;
  if (idx >= total) return;
  int i = (int)(idx & 63);
  size_t p = idx >> 6;
  int h = (int)(p % nheads); p /= nheads;
  int t = (int)(p & 2047);
  int b = (int)(p >> 11);

  const float* s = src + (((size_t)b * 2048 + t) * nheads + h) * 128 + 2 * i;
  float xr = s[0], xi = s[1];
  float orr = xr, oi = xi;
  if (do_rope) {
    float theta = __expf((float)(-2 * i) * 0.07195578814f);  // ln(10000)/128
    float ang = (float)t * theta;
    float sn, cs;
    __sincosf(ang, &sn, &cs);
    orr = xr * cs - xi * sn;
    oi  = xr * sn + xi * cs;
  }
  __bf16* d = dst + (((size_t)b * nheads + h) * 2048 + t) * 128 + 2 * i;
  d[0] = f2bf(orr);
  d[1] = f2bf(oi);
}

// ---------------------------------------------------------------------------
// Flash attention: block = (b, head, 64-row q tile); 128 threads = 4 waves.
// Q frags pinned in VGPRs; K tile [key][d]; V tile transposed [d][key];
// P re-laid out C->A through per-wave LDS staging; bf16 output (A of out-proj).
// ---------------------------------------------------------------------------
__global__ __launch_bounds__(128) void attn_fa_wmma(
    const __bf16* __restrict__ Qb, const __bf16* __restrict__ Kb,
    const __bf16* __restrict__ Vb, __bf16* __restrict__ Out) {
  __shared__ __attribute__((aligned(16))) __bf16 Kt[64 * 136];
  __shared__ __attribute__((aligned(16))) __bf16 Vt[128 * 72];
  __shared__ __attribute__((aligned(16))) __bf16 Ps[4][16 * 72];

  int bid = blockIdx.x;
  int qb  = bid & 31;
  int h   = (bid >> 5) & 31;
  int b   = bid >> 10;
  int kvh = h >> 2;

  const __bf16* Qp = Qb + (((size_t)b * 32 + h) * 2048 + (size_t)qb * 64) * 128;
  const __bf16* Kp = Kb + ((size_t)b * 8 + kvh) * 2048 * 128;
  const __bf16* Vp = Vb + ((size_t)b * 8 + kvh) * 2048 * 128;

  const int tid  = threadIdx.x;
  const int lane = tid & 31;
  const int wave = tid >> 5;
  const int hh   = lane >> 4;
  const int nl   = lane & 15;

  v16bf qf[4];
#pragma unroll
  for (int s = 0; s < 4; ++s) {
    const __bf16* qr = Qp + (size_t)(wave * 16 + nl) * 128 + s * 32 + hh * 8;
    v8bf lo = *(const v8bf*)qr;
    v8bf hi = *(const v8bf*)(qr + 16);
    qf[s] = cat8(lo, hi);
  }

  v8f o[8];
#pragma unroll
  for (int dt = 0; dt < 8; ++dt) o[dt] = vzero8();
  float mrow[8], lrow[8];
#pragma unroll
  for (int r = 0; r < 8; ++r) { mrow[r] = -1e30f; lrow[r] = 0.f; }

  const float scale = 0.08838834764831845f;  // 1/sqrt(128)
  const int nkt = qb + 1;                    // causal tile skip (block-uniform)

  for (int kt = 0; kt < nkt; ++kt) {
    __syncthreads();
    const __bf16* Ksrc = Kp + (size_t)kt * 64 * 128;
    const __bf16* Vsrc = Vp + (size_t)kt * 64 * 128;
#pragma unroll
    for (int i = 0; i < 8; ++i) {
      int idx = tid + 128 * i;
      int key = idx >> 4;
      int c8  = (idx & 15) << 3;
      v8bf kv = *(const v8bf*)(Ksrc + (size_t)key * 128 + c8);
      *(v8bf*)&Kt[key * 136 + c8] = kv;
      v8bf vv = *(const v8bf*)(Vsrc + (size_t)key * 128 + c8);
#pragma unroll
      for (int j = 0; j < 8; ++j) Vt[(c8 + j) * 72 + key] = vv[j];
    }
    __syncthreads();

    v8f sc[4];
#pragma unroll
    for (int nt = 0; nt < 4; ++nt) sc[nt] = vzero8();
#pragma unroll
    for (int nt = 0; nt < 4; ++nt)
#pragma unroll
      for (int s = 0; s < 4; ++s) {
        v16bf kf = frag_ld(&Kt[(nt * 16 + nl) * 136 + s * 32], hh);
        sc[nt] = WMMA_BF16(qf[s], kf, sc[nt]);
      }

    int kbase = kt * 64;
#pragma unroll
    for (int r = 0; r < 8; ++r) {
      int qrow = qb * 64 + wave * 16 + 8 * hh + r;
      float mx = mrow[r];
#pragma unroll
      for (int nt = 0; nt < 4; ++nt) {
        int key = kbase + nt * 16 + nl;
        float sv = sc[nt][r] * scale;
        sv = (key <= qrow) ? sv : -1e30f;
        sc[nt][r] = sv;
        mx = fmaxf(mx, sv);
      }
#pragma unroll
      for (int msk = 8; msk >= 1; msk >>= 1)
        mx = fmaxf(mx, __shfl_xor(mx, msk, 32));
      float corr = __expf(mrow[r] - mx);
      mrow[r] = mx;
      float ps = 0.f;
#pragma unroll
      for (int nt = 0; nt < 4; ++nt) {
        float pv = __expf(sc[nt][r] - mx);
        sc[nt][r] = pv;
        ps += pv;
      }
#pragma unroll
      for (int msk = 8; msk >= 1; msk >>= 1)
        ps += __shfl_xor(ps, msk, 32);
      lrow[r] = lrow[r] * corr + ps;
#pragma unroll
      for (int dt = 0; dt < 8; ++dt) o[dt][r] *= corr;
    }

    __bf16* pw = &Ps[wave][0];
#pragma unroll
    for (int r = 0; r < 8; ++r)
#pragma unroll
      for (int nt = 0; nt < 4; ++nt)
        pw[(8 * hh + r) * 72 + nt * 16 + nl] = f2bf(sc[nt][r]);
    __syncthreads();

#pragma unroll
    for (int kk = 0; kk < 2; ++kk) {
      v16bf pa = frag_ld(&pw[nl * 72 + kk * 32], hh);
#pragma unroll
      for (int dt = 0; dt < 8; ++dt) {
        v16bf vf = frag_ld(&Vt[(dt * 16 + nl) * 72 + kk * 32], hh);
        o[dt] = WMMA_BF16(pa, vf, o[dt]);
      }
    }
  }

#pragma unroll
  for (int r = 0; r < 8; ++r) {
    float inv = 1.f / lrow[r];
    int qrow = qb * 64 + wave * 16 + 8 * hh + r;
    __bf16* orow = Out + ((size_t)b * 2048 + qrow) * 4096 + h * 128;
#pragma unroll
    for (int dt = 0; dt < 8; ++dt)
      orow[dt * 16 + nl] = f2bf(o[dt][r] * inv);
  }
}

// ---------------------------------------------------------------------------
// Orchestration. Workspace (~268 MB):
//   xbf 32M | wqT 32M | wkT 8M | wvT 8M | woT 32M | Qf 64M | Kf 16M | Vf 16M
//   | Qbf 32M | Kbf 8M | Vbf 8M ; attention bf16 output reuses the xbf region.
// ---------------------------------------------------------------------------
extern "C" void kernel_launch(void* const* d_in, const int* in_sizes, int n_in,
                              void* d_out, int out_size, void* d_ws, size_t ws_size,
                              hipStream_t stream) {
  const float* x  = (const float*)d_in[0];
  const float* wq = (const float*)d_in[1];
  const float* wk = (const float*)d_in[2];
  const float* wv = (const float*)d_in[3];
  const float* wo = (const float*)d_in[4];
  float* out = (float*)d_out;

  const size_t NQ = (size_t)4096 * 4096;
  const size_t NK = (size_t)4096 * 1024;

  __bf16* xbf = (__bf16*)d_ws;
  __bf16* wqT = xbf + NQ;
  __bf16* wkT = wqT + NQ;
  __bf16* wvT = wkT + NK;
  __bf16* woT = wvT + NK;
  float*  Qf  = (float*)(woT + NQ);
  float*  Kf  = Qf + NQ;
  float*  Vf  = Kf + NK;
  __bf16* Qb  = (__bf16*)(Vf + NK);
  __bf16* Kb  = Qb + NQ;
  __bf16* Vb  = Kb + NK;
  __bf16* Ab  = xbf;  // attention bf16 output (xbf dead after QKV GEMMs)

  dim3 blk(256);
  // One-time format conversion
  cast_bf16<<<(unsigned)(NQ / 4 / 256), blk, 0, stream>>>(x, xbf, (unsigned)(NQ / 4));
  transpose_cast<<<dim3(4096 / 32, 4096 / 32), blk, 0, stream>>>(wq, wqT, 4096, 4096);
  transpose_cast<<<dim3(1024 / 32, 4096 / 32), blk, 0, stream>>>(wk, wkT, 4096, 1024);
  transpose_cast<<<dim3(1024 / 32, 4096 / 32), blk, 0, stream>>>(wv, wvT, 4096, 1024);
  transpose_cast<<<dim3(4096 / 32, 4096 / 32), blk, 0, stream>>>(wo, woT, 4096, 4096);

  // QKV projections (TDM-fed bf16 WMMA)
  gemm_bf16_tdm<<<dim3(4096 / 128, 4096 / 128), blk, 0, stream>>>(xbf, wqT, Qf, 4096, 4096);
  gemm_bf16_tdm<<<dim3(1024 / 128, 4096 / 128), blk, 0, stream>>>(xbf, wkT, Kf, 4096, 1024);
  gemm_bf16_tdm<<<dim3(1024 / 128, 4096 / 128), blk, 0, stream>>>(xbf, wvT, Vf, 4096, 1024);

  // RoPE + pack to [b][head][t][d] bf16
  rope_pack<<<(unsigned)((size_t)2 * 2048 * 32 * 64 / 256), blk, 0, stream>>>(Qf, Qb, 32, 1);
  rope_pack<<<(unsigned)((size_t)2 * 2048 * 8 * 64 / 256), blk, 0, stream>>>(Kf, Kb, 8, 1);
  rope_pack<<<(unsigned)((size_t)2 * 2048 * 8 * 64 / 256), blk, 0, stream>>>(Vf, Vb, 8, 0);

  // Attention: 2 batches x 32 heads x 32 q-tiles
  attn_fa_wmma<<<2 * 32 * 32, dim3(128), 0, stream>>>(Qb, Kb, Vb, Ab);

  // Output projection
  gemm_bf16_tdm<<<dim3(4096 / 128, 4096 / 128), blk, 0, stream>>>(Ab, woT, out, 4096, 4096);
}
